// TensorBoard_82600811036959
// MI455X (gfx1250) — matrix operand order repaired and checked by reading, not compiled, not run on hardware
//
#include <hip/hip_runtime.h>

typedef _Float16 v16h __attribute__((ext_vector_type(16)));
typedef _Float16 v8h  __attribute__((ext_vector_type(8)));
typedef float    v8f  __attribute__((ext_vector_type(8)));

#define GOH 19

static __device__ __forceinline__ v8f wmma_f16(v16h a, v16h b, v8f c) {
    return __builtin_amdgcn_wmma_f32_16x16x32_f16(false, a, false, b, (short)0, c, false, false);
}

// One wave = one 19x19 board, 4 waves/block. All four stencils nbr(X)=A*X+X*A
// run on the matrix pipes (v_wmma_f32_16x16x32_f16, exact for small ints);
// per-cell values are extracted via identity-matrix WMMAs (I*X) instead of
// scalar LDS reads. ADJ/IDM operand planes are built once per block in LDS.
__global__ __launch_bounds__(128) void go_features_kernel(
    const float* __restrict__ stones,   // (B,2,19,19) f32
    const int*   __restrict__ cp_arr,   // (B,) i32
    const int*   __restrict__ ko,       // (B,2) i32
    float*       __restrict__ out,      // (B,5,19,19) f32
    int nBoards)
{
    // per-wave planes: 0=E, 1=E^T, 2=C, 3=C^T, 4=O(->W), 5=O^T(->W^T)
    __shared__ __align__(16) _Float16 P[4][6][32][32];
    __shared__ __align__(16) _Float16 ADJ[32][32];   // path adjacency (symmetric)
    __shared__ __align__(16) _Float16 IDM[32][32];   // identity (symmetric)

    const int tid = threadIdx.x;
    const int w   = tid >> 5;
    const int l   = tid & 31;
    int b = blockIdx.x * 4 + w;
    const bool active = (b < nBoards);
    if (b >= nBoards) b = nBoards - 1;   // clamp; stores guarded by `active`

    const int  lm = l & 15;
    const bool hi = (l >= 16);

    // ---- Phase 0: zero per-wave planes (b128); build shared ADJ/IDM ----
    {
        uint4 z = make_uint4(0u, 0u, 0u, 0u);
        uint4* pz = (uint4*)&P[w][0][0][0];
        #pragma unroll
        for (int i = 0; i < 24; ++i) pz[l + 32 * i] = z;
        if (tid < 32) {                    // thread r owns row r of both planes
            int r = tid;
            uint4* az = (uint4*)&ADJ[r][0];
            uint4* iz = (uint4*)&IDM[r][0];
            #pragma unroll
            for (int i = 0; i < 4; ++i) { az[i] = z; iz[i] = z; }
            const _Float16 one = (_Float16)1.0f;
            if (r > 0)  ADJ[r][r - 1] = one;
            if (r < 31) ADJ[r][r + 1] = one;
            IDM[r][r] = one;
        }
    }
    __syncthreads();

    // ---- Per-board scalars ----
    const long sb = (long)b * (2 * GOH * GOH);
    const long ob = (long)b * (5 * GOH * GOH);
    const int  cp = cp_arr[b];
    const float turn = (float)cp;
    const int k0 = ko[2 * b], k1 = ko[2 * b + 1];
    const bool has_ko = (k0 >= 0);
    const int rr = k0 < 0 ? 0 : (k0 > 18 ? 18 : k0);
    const int cc = k1 < 0 ? 0 : (k1 > 18 ? 18 : k1);

    // ---- Phase 1: coalesced load, fill planes, stream cur/opp/turn ----
    for (int i = l; i < GOH * GOH; i += 32) {
        float s0 = stones[sb + i];
        float s1 = stones[sb + GOH * GOH + i];
        float cur = (cp == 0) ? s0 : s1;
        float opp = (cp == 0) ? s1 : s0;
        float emp = 1.0f - s0 - s1;
        int r = i / GOH, c = i - r * GOH;
        _Float16 he = (_Float16)emp, hc = (_Float16)cur, ho = (_Float16)opp;
        P[w][0][r][c] = he;  P[w][1][c][r] = he;
        P[w][2][r][c] = hc;  P[w][3][c][r] = hc;
        P[w][4][r][c] = ho;  P[w][5][c][r] = ho;
        if (active) {
            float* o = out + ob + i;
            o[0]             = cur;
            o[GOH * GOH]     = opp;
            o[4 * GOH * GOH] = turn;
        }
    }
    __syncthreads();

    // ---- Operand loaders: two ds_load_b128 each (layout-exact, 16B aligned) ----
    auto loadAp = [&](const _Float16 (*plane)[32], int I) -> v16h {
        const int m  = 16 * I + lm;
        const int kb = hi ? 8 : 0;
        v8h lo = *(const v8h*)&plane[m][kb];
        v8h hp = *(const v8h*)&plane[m][16 + kb];
        v16h r;
        #pragma unroll
        for (int e = 0; e < 8; ++e) { r[e] = lo[e]; r[8 + e] = hp[e]; }
        return r;
    };
    auto loadBp = [&](const _Float16 (*planeT)[32], int J) -> v16h {
        const int n  = 16 * J + lm;
        const int kk = hi ? 8 : 0;
        v8h lo = *(const v8h*)&planeT[n][kk];
        v8h hp = *(const v8h*)&planeT[n][16 + kk];
        v16h r;
        #pragma unroll
        for (int e = 0; e < 8; ++e) { r[e] = lo[e]; r[8 + e] = hp[e]; }
        return r;
    };

    // Resident constant operands (ADJ symmetric => same plane for A and B side)
    v16h adjA[2] = { loadAp(ADJ, 0), loadAp(ADJ, 1) };
    v16h adjB[2] = { loadBp(ADJ, 0), loadBp(ADJ, 1) };
    v16h idA [2] = { loadAp(IDM, 0), loadAp(IDM, 1) };

    // ---- Phase 2: nbr(O) + O values via WMMA; derive W, overwrite O planes ----
    {
        v16h Oa[2] = { loadAp(P[w][4], 0), loadAp(P[w][4], 1) };
        v16h Ob[2] = { loadBp(P[w][5], 0), loadBp(P[w][5], 1) };
        v8f nO[4], vO[4];
        #pragma unroll
        for (int I = 0; I < 2; ++I)
        #pragma unroll
        for (int J = 0; J < 2; ++J) {
            v8f c = {};
            c = wmma_f16(adjA[I], Ob[J], c);     // A*O
            c = wmma_f16(Oa[I], adjB[J], c);     // O*A
            nO[2 * I + J] = c;
            v8f z = {};
            vO[2 * I + J] = wmma_f16(idA[I], Ob[J], z);  // I*O = O values
        }
        // W stores depend on the operand loads through the WMMA chain and LDS
        // is in-order within a wave, so no barrier is needed before overwrite.
        #pragma unroll
        for (int t = 0; t < 4; ++t) {
            const int I = t >> 1, J = t & 1;
            #pragma unroll
            for (int v = 0; v < 8; ++v) {
                int r = 16 * I + v + (hi ? 8 : 0);
                int c = 16 * J + lm;
                if (r < GOH && c < GOH) {
                    bool wk = (vO[t][v] > 0.5f) && (nO[t][v] == 1.0f);
                    _Float16 hw = wk ? (_Float16)1.0f : (_Float16)0.0f;
                    P[w][4][r][c] = hw;          // W   (padding already zero)
                    P[w][5][c][r] = hw;          // W^T
                }
            }
        }
    }
    __syncthreads();

    // ---- Phase 3: nbr(E), nbr(C), nbr(W), E values via WMMA; finalize ----
    #pragma unroll
    for (int I = 0; I < 2; ++I)
    #pragma unroll
    for (int J = 0; J < 2; ++J) {
        v16h Ea = loadAp(P[w][0], I), Eb = loadBp(P[w][1], J);
        v16h Ca = loadAp(P[w][2], I), Cb = loadBp(P[w][3], J);
        v16h Wa = loadAp(P[w][4], I), Wb = loadBp(P[w][5], J);

        v8f aE = {};
        aE = wmma_f16(adjA[I], Eb, aE);
        aE = wmma_f16(Ea, adjB[J], aE);
        v8f aC = {};
        aC = wmma_f16(adjA[I], Cb, aC);
        aC = wmma_f16(Ca, adjB[J], aC);
        v8f aW = {};
        aW = wmma_f16(adjA[I], Wb, aW);
        aW = wmma_f16(Wa, adjB[J], aW);
        v8f zz = {};
        v8f eV = wmma_f16(idA[I], Eb, zz);       // I*E = empty values

        const int cN = 16 * J + lm;
        const int r0 = 16 * I + (hi ? 8 : 0);
        float* o2 = out + ob + 2 * GOH * GOH + r0 * GOH + cN;
        float* o3 = out + ob + 3 * GOH * GOH + r0 * GOH + cN;
        #pragma unroll
        for (int v = 0; v < 8; ++v) {
            int r = r0 + v;
            if (active && r < GOH && cN < GOH) {
                float e = eV[v];
                float s = aE[v] + aW[v];         // both >= 0
                float legal = (e > 0.5f && s > 0.0f) ? 1.0f : 0.0f;
                o2[v * GOH] = legal;             // imm-offset stores
                o3[v * GOH] = aC[v] * e;         // cur_lib = nbr(cur)*empty
            }
        }
    }

    // ---- Ko fixup: single store; same-wave stores are ordered ----
    if (active && has_ko && l == 0) {
        out[ob + 2 * GOH * GOH + rr * GOH + cc] = 0.0f;
    }
}

extern "C" void kernel_launch(void* const* d_in, const int* in_sizes, int n_in,
                              void* d_out, int out_size, void* d_ws, size_t ws_size,
                              hipStream_t stream) {
    (void)n_in; (void)out_size; (void)d_ws; (void)ws_size;
    const float* stones = (const float*)d_in[0];
    const int*   cp     = (const int*)d_in[1];
    const int*   ko     = (const int*)d_in[2];
    float*       out    = (float*)d_out;
    const int nBoards = in_sizes[1];               // current_player count == B
    const int blocks  = (nBoards + 3) / 4;         // 4 waves (boards) per block
    go_features_kernel<<<blocks, 128, 0, stream>>>(stones, cp, ko, out, nBoards);
}
